// CasualSelfAttention_50465865728002
// MI455X (gfx1250) — matrix operand order repaired and checked by reading, not compiled, not run on hardware
//
#include <hip/hip_runtime.h>

typedef _Float16 h16;
typedef __attribute__((ext_vector_type(16))) _Float16 v16h;
typedef __attribute__((ext_vector_type(8)))  float    v8f;
typedef __attribute__((ext_vector_type(4)))  unsigned int u32x4;
typedef __attribute__((ext_vector_type(8)))  int i32x8;
typedef __attribute__((ext_vector_type(4)))  int i32x4;

constexpr int Bz = 4, Tz = 2048, Cz = 1024, NHz = 16, Hd = 64;
constexpr int C3 = 3 * Cz;

static __device__ __forceinline__ v8f wmma32(v16h a, v16h b, v8f c) {
  // D = A(16x32 f16) * B(32x16 f16) + C(16x16 f32)
  return __builtin_amdgcn_wmma_f32_16x16x32_f16(false, a, false, b,
                                                (short)0, c, false, false);
}

// A-fragment (16x32 f16) element -> K index.
// lanes 0-15: e0..15 -> K=e; lanes 16-31: K=16+e.
static __device__ __forceinline__ int a_k(int e, int half) {
  return e + 16 * half;
}

// ---------------------------------------------------------------------------
// TDM: async 2D tile load (2-byte elements) global -> LDS.
// Builds D# groups per CDNA5 ISA ch.8 and issues TENSOR_LOAD_TO_LDS.
//   dim0/stride0 in elements along fastest axis, dim1 = #rows in tensor,
//   tile0 x tile1 = tile shape (elements x rows). LDS result is row-major
//   tile0-contiguous.  (6-arg builtin: g0, g1, g2, g3, g4, cpol)
// ---------------------------------------------------------------------------
static __device__ __forceinline__ void tdm_load_2d_f16(
    unsigned lds_off, const void* gaddr, unsigned dim0, unsigned dim1,
    unsigned stride0, unsigned tile0, unsigned tile1) {
  const unsigned long long ga = (unsigned long long)gaddr;
  u32x4 g0;
  g0[0] = 1u;                                  // count=1, user descriptor
  g0[1] = lds_off;                             // lds_addr (bytes)
  g0[2] = (unsigned)ga;                        // global_addr[31:0]
  g0[3] = (unsigned)(ga >> 32) | (2u << 30);   // global_addr[56:32] | type=2
  i32x8 g1;
  g1[0] = 0x10000;                             // data_size=1 (2B), no flags
  g1[1] = (int)((dim0 & 0xFFFFu) << 16);       // tensor_dim0 lo16 (<<16)
  g1[2] = (int)((dim0 >> 16) | ((dim1 & 0xFFFFu) << 16));
  g1[3] = (int)((dim1 >> 16) | (tile0 << 16)); // tensor_dim1 hi | tile_dim0
  g1[4] = (int)tile1;                          // tile_dim1, tile_dim2=0
  g1[5] = (int)stride0;                        // tensor_dim0_stride lo32
  g1[6] = 0;                                   // stride0 hi | dim1_stride lo
  g1[7] = 0;
  const i32x4 z4 = {0, 0, 0, 0};               // groups 2/3 unused (<=2D)
  const i32x8 z8 = {0, 0, 0, 0, 0, 0, 0, 0};
  __builtin_amdgcn_tensor_load_to_lds(g0, g1, z4, z4, z8, 0);
}

static __device__ __forceinline__ unsigned lds_offset(const void* p) {
  // Generic LDS address keeps the LDS byte offset in addr[31:0].
  return (unsigned)(unsigned long long)p;
}

// ---------------------------------------------------------------------------
// Kernel 1: qkv = x @ W_attn + b_attn -> Q,K row-major [B,NH,T,H] (f16),
//           V transposed [B,NH,H,T] (f16). One wave per 16x16 tile.
// ---------------------------------------------------------------------------
__global__ void qkv_kernel(const float* __restrict__ x,
                           const float* __restrict__ W,
                           const float* __restrict__ bias,
                           h16* __restrict__ q, h16* __restrict__ k,
                           h16* __restrict__ vT) {
  const int lane = threadIdx.x & 31;
  const int warp = threadIdx.x >> 5;
  const int l16 = lane & 15, half = lane >> 4;
  const int NT = C3 / 16;  // 192 column tiles
  const int tile = blockIdx.x * 4 + warp;
  const int mbase = (tile / NT) * 16;
  const int nbase = (tile % NT) * 16;

  v8f acc = {};
  for (int kb = 0; kb < Cz; kb += 32) {
    v16h a, b;
    const float* xr = x + (size_t)(mbase + l16) * Cz + kb;
#pragma unroll
    for (int e = 0; e < 16; ++e) a[e] = (h16)xr[a_k(e, half)];
    const float* wc = W + (size_t)kb * C3 + nbase + l16;
    if (kb + 32 < Cz)  // stream next weight panel toward L2
      __builtin_prefetch(wc + (size_t)32 * C3, 0, 1);
#pragma unroll
    for (int e = 0; e < 16; ++e) b[e] = (h16)wc[(size_t)(e + 16 * half) * C3];
    acc = wmma32(a, b, acc);
  }

  const int col = nbase + l16;           // 0..3C-1
  const float bv = bias[col];
  const int which = col / Cz;            // 0=q 1=k 2=v
  const int cin = col % Cz;
  const int head = cin / Hd;
  const int hd = cin % Hd;
  const int bb = mbase / Tz;             // tiles never cross batch (16 | T)
  const int tb = mbase % Tz;
#pragma unroll
  for (int r = 0; r < 8; ++r) {
    const int t = tb + r + 8 * half;
    const h16 hv = (h16)(acc[r] + bv);
    if (which == 0)
      q[(size_t)((bb * NHz + head) * Tz + t) * Hd + hd] = hv;
    else if (which == 1)
      k[(size_t)((bb * NHz + head) * Tz + t) * Hd + hd] = hv;
    else
      vT[(size_t)((bb * NHz + head) * Hd + hd) * Tz + t] = hv;
  }
}

// ---------------------------------------------------------------------------
// Kernel 2: flash attention, one wave per (b, head, 16-query tile).
// K/V^T tiles are DMA'd into LDS by the Tensor Data Mover (double-buffered,
// tracked with TENSORcnt); WMMA fragments are built from LDS.
// ---------------------------------------------------------------------------
__global__ void attn_kernel(const h16* __restrict__ q,
                            const h16* __restrict__ k,
                            const h16* __restrict__ vT,
                            h16* __restrict__ y) {
  __shared__ h16 kt[2][32 * Hd];   // 2 x (32 keys x 64 d), 8 KB
  __shared__ h16 vt[2][Hd * 32];   // 2 x (64 d   x 32 keys), 8 KB
  __shared__ h16 pt[16 * 32];      // P tile, 1 KB

  const int lane = threadIdx.x & 31;
  const int l16 = lane & 15, half = lane >> 4;
  const int qt = blockIdx.x;       // query tile 0..127
  const int h = blockIdx.y, b = blockIdx.z;
  const int qbase = qt * 16;
  const size_t bh = (size_t)(b * NHz + h);
  const h16* Q = q + bh * Tz * Hd;
  const h16* K = k + bh * Tz * Hd;
  const h16* V = vT + bh * Hd * Tz;

  // Q A-fragments, head-dim chunks d=0..31 and d=32..63 (reused all loop)
  v16h aq0, aq1;
  {
    const h16* qr = Q + (size_t)(qbase + l16) * Hd;
#pragma unroll
    for (int e = 0; e < 16; ++e) {
      const int kk = a_k(e, half);
      aq0[e] = qr[kk];
      aq1[e] = qr[32 + kk];
    }
  }

  float m[8], l[8];
  v8f o[4] = {v8f{}, v8f{}, v8f{}, v8f{}};
#pragma unroll
  for (int r = 0; r < 8; ++r) { m[r] = -1e30f; l[r] = 0.f; }
  const float scale = 0.125f;  // 1/sqrt(64)

  const int nb = (qbase + 16 + 31) / 32;  // # of 32-key blocks (causal)

  // Prologue: DMA block 0 into buffer 0.
  //   K tile:  tensor 64 x 2048 (row=key), tile 64 x 32  -> kt[key][d]
  //   V^T tile: tensor 2048 x 64 (row=d),  tile 32 x 64  -> vt[d][key]
  tdm_load_2d_f16(lds_offset(&kt[0][0]), K, Hd, Tz, Hd, Hd, 32);
  tdm_load_2d_f16(lds_offset(&vt[0][0]), V, Tz, Hd, Tz, 32, Hd);

  for (int ib = 0; ib < nb; ++ib) {
    const int kb = ib * 32;
    const int cur = ib & 1;
    if (ib + 1 < nb) {  // DMA next block into the other buffer
      const int kn = kb + 32;
      tdm_load_2d_f16(lds_offset(&kt[cur ^ 1][0]), K + (size_t)kn * Hd,
                      Hd, Tz, Hd, Hd, 32);
      tdm_load_2d_f16(lds_offset(&vt[cur ^ 1][0]), V + kn,
                      Tz, Hd, Tz, 32, Hd);
      __builtin_amdgcn_s_wait_tensorcnt(2);  // current buffer complete
    } else {
      __builtin_amdgcn_s_wait_tensorcnt(0);
    }
    const h16* kl = &kt[cur][0];
    const h16* vl = &vt[cur][0];

    // ---- S = Q K^T over 32 keys (two 16-key sub-tiles) from LDS ----
    v16h b0, b1;
#pragma unroll
    for (int e = 0; e < 16; ++e) {
      const int dd = e + 16 * half;  // d = 0..31
      b0[e] = kl[l16 * Hd + dd];
      b1[e] = kl[(16 + l16) * Hd + dd];
    }
    v8f s0 = {}, s1 = {};
    s0 = wmma32(aq0, b0, s0);
    s1 = wmma32(aq0, b1, s1);
#pragma unroll
    for (int e = 0; e < 16; ++e) {
      const int dd = 32 + e + 16 * half;  // d = 32..63
      b0[e] = kl[l16 * Hd + dd];
      b1[e] = kl[(16 + l16) * Hd + dd];
    }
    s0 = wmma32(aq1, b0, s0);
    s1 = wmma32(aq1, b1, s1);

    // ---- causal mask + online softmax (rows live across 16 lanes) ----
    const int kg0 = kb + l16, kg1 = kb + 16 + l16;
#pragma unroll
    for (int r = 0; r < 8; ++r) {
      const int qg = qbase + r + 8 * half;  // global query row
      float e0 = (kg0 <= qg) ? s0[r] * scale : -1e30f;
      float e1 = (kg1 <= qg) ? s1[r] * scale : -1e30f;
      float mx = fmaxf(e0, e1);
      mx = fmaxf(mx, __shfl_xor(mx, 1));
      mx = fmaxf(mx, __shfl_xor(mx, 2));
      mx = fmaxf(mx, __shfl_xor(mx, 4));
      mx = fmaxf(mx, __shfl_xor(mx, 8));
      const float mnew = fmaxf(m[r], mx);
      const float alpha = __expf(m[r] - mnew);
      const float p0 = __expf(e0 - mnew);
      const float p1 = __expf(e1 - mnew);
      float ps = p0 + p1;
      ps += __shfl_xor(ps, 1);
      ps += __shfl_xor(ps, 2);
      ps += __shfl_xor(ps, 4);
      ps += __shfl_xor(ps, 8);
      l[r] = l[r] * alpha + ps;
      m[r] = mnew;
      o[0][r] *= alpha; o[1][r] *= alpha; o[2][r] *= alpha; o[3][r] *= alpha;
      // stage P row-major 16x32 (D-layout -> memory)
      const int row = r + 8 * half;
      pt[row * 32 + l16] = (h16)p0;
      pt[row * 32 + 16 + l16] = (h16)p1;
    }

    // ---- reload P as A-fragment (same-wave LDS is in-order) ----
    v16h ap;
#pragma unroll
    for (int e = 0; e < 16; ++e) ap[e] = pt[l16 * 32 + a_k(e, half)];

    // ---- O += P V : vt[d][key] makes B-fragments contiguous ----
#pragma unroll
    for (int n = 0; n < 4; ++n) {
      v16h bv;
      const h16* vr = vl + (size_t)(n * 16 + l16) * 32 + 16 * half;
#pragma unroll
      for (int e = 0; e < 16; ++e) bv[e] = vr[e];
      o[n] = wmma32(ap, bv, o[n]);
    }
  }

  // ---- epilogue: divide by row-sum, store f16 y[B,T,C] ----
#pragma unroll
  for (int r = 0; r < 8; ++r) {
    const int t = qbase + r + 8 * half;
    const float inv = 1.0f / l[r];
#pragma unroll
    for (int n = 0; n < 4; ++n)
      y[(size_t)(b * Tz + t) * Cz + h * Hd + n * 16 + l16] =
          (h16)(o[n][r] * inv);
  }
}

// ---------------------------------------------------------------------------
// Kernel 3: out = y @ W_proj + b_proj  (fp32 output)
// ---------------------------------------------------------------------------
__global__ void proj_kernel(const h16* __restrict__ y,
                            const float* __restrict__ W,
                            const float* __restrict__ bias,
                            float* __restrict__ out) {
  const int lane = threadIdx.x & 31;
  const int warp = threadIdx.x >> 5;
  const int l16 = lane & 15, half = lane >> 4;
  const int NT = Cz / 16;  // 64 column tiles
  const int tile = blockIdx.x * 4 + warp;
  const int mbase = (tile / NT) * 16;
  const int nbase = (tile % NT) * 16;

  v8f acc = {};
  for (int kb = 0; kb < Cz; kb += 32) {
    v16h a, b;
    const h16* yr = y + (size_t)(mbase + l16) * Cz + kb;
#pragma unroll
    for (int e = 0; e < 16; ++e) a[e] = yr[a_k(e, half)];
    const float* wc = W + (size_t)kb * Cz + nbase + l16;
    if (kb + 32 < Cz)
      __builtin_prefetch(wc + (size_t)32 * Cz, 0, 1);
#pragma unroll
    for (int e = 0; e < 16; ++e) b[e] = (h16)wc[(size_t)(e + 16 * half) * Cz];
    acc = wmma32(a, b, acc);
  }
  const float bv = bias[nbase + l16];
#pragma unroll
  for (int r = 0; r < 8; ++r) {
    const int gm = mbase + r + 8 * half;
    out[(size_t)gm * Cz + nbase + l16] = acc[r] + bv;
  }
}

// ---------------------------------------------------------------------------
extern "C" void kernel_launch(void* const* d_in, const int* in_sizes, int n_in,
                              void* d_out, int out_size, void* d_ws,
                              size_t ws_size, hipStream_t stream) {
  (void)in_sizes; (void)n_in; (void)out_size; (void)ws_size;
  const float* x      = (const float*)d_in[0];
  const float* W_attn = (const float*)d_in[1];
  const float* b_attn = (const float*)d_in[2];
  const float* W_proj = (const float*)d_in[3];
  const float* b_proj = (const float*)d_in[4];
  float* out = (float*)d_out;

  const size_t n1 = (size_t)Bz * NHz * Tz * Hd;  // 8,388,608 elements
  h16* q  = (h16*)d_ws;
  h16* k  = q + n1;
  h16* vT = k + n1;
  h16* y  = vT + n1;  // total 4 * 16 MB of f16 scratch

  // QKV GEMM: (B*T/16)*(3C/16) = 512*192 = 98304 wave-tiles, 4 waves/block
  qkv_kernel<<<98304 / 4, 128, 0, stream>>>(x, W_attn, b_attn, q, k, vT);
  // Attention: 128 query tiles x 16 heads x 4 batches, 1 wave/block
  attn_kernel<<<dim3(Tz / 16, NHz, Bz), 32, 0, stream>>>(q, k, vT, y);
  // Output projection: 512*64 = 32768 wave-tiles, 4 waves/block
  proj_kernel<<<32768 / 4, 128, 0, stream>>>(y, W_proj, b_proj, out);
}
